// InceptionResNet_82643760709720
// MI455X (gfx1250) — matrix operand order repaired and checked by reading, not compiled, not run on hardware
//
#include <hip/hip_runtime.h>

// ---------- types for WMMA fragments ----------
typedef __bf16        v16bf __attribute__((ext_vector_type(16)));
typedef float         v8f   __attribute__((ext_vector_type(8)));
typedef unsigned int  v4u   __attribute__((ext_vector_type(4)));

union ABFrag { v4u u[2]; v16bf bf; };

// fp32 -> bf16 (round to nearest even), raw u16 storage
static __device__ __forceinline__ unsigned short f2bf(float f) {
    unsigned int u = __float_as_uint(f);
    u += 0x7fffu + ((u >> 16) & 1u);
    return (unsigned short)(u >> 16);
}

// ---------- prep kernels ----------
// x [n*128] f32 -> xpad [(n+1)*128] bf16, final row zeroed ("missing neighbor" row)
__global__ void k_cvt_x(const float* __restrict__ x, unsigned short* __restrict__ xp,
                        int total, int padtotal) {
    int i = blockIdx.x * 256 + threadIdx.x;
    if (i < padtotal) {
        float v = (i < total) ? x[i] : 0.0f;
        xp[i] = f2bf(v);
    }
}

// W [KK, CIN, 64] f32 -> Wt [KK, 64, CIN] bf16 (output-channel-major so a WMMA
// B fragment = two runs of 8 contiguous bf16)
__global__ void k_cvt_w(const float* __restrict__ W, unsigned short* __restrict__ Wt,
                        int KK, int CIN) {
    int i = blockIdx.x * 256 + threadIdx.x;
    int tot = KK * CIN * 64;
    if (i < tot) {
        int k   = i / (64 * CIN);
        int rem = i - k * (64 * CIN);
        int co  = rem / CIN;
        int ci  = rem - co * CIN;
        Wt[i] = f2bf(W[(k * CIN + ci) * 64 + co]);
    }
}

// zero the pad row (row N, 64 channels) of the three intermediate bf16 buffers
__global__ void k_zero_pad(unsigned short* a, unsigned short* b, unsigned short* c) {
    int i = threadIdx.x;           // 64 threads
    a[i] = 0; b[i] = 0; c[i] = 0;
}

// ---------- core gather-GEMM kernel ----------
// Block = 256 threads = 8 waves, 128 points. Wave w computes a 16x64 output tile
// for points [blockIdx.x*128 + w*16, +16). K = KOFF kernel offsets x CIN channels.
// Weights for KC offsets at a time are staged in LDS (all waves share them);
// gathered A rows are double-buffered in registers (prefetch k+1 during k).
// BF16OUT: out = bf16(relu(acc + bias)) into obf [(n+1)*64]  (pad row stays zero)
// else   : out = acc + bias + xres     into ofl [n*128] at column offset col0
template <int CIN, int KOFF, int KC, bool GATHER, bool BF16OUT>
__global__ __launch_bounds__(256)
void k_spconv(const unsigned short* __restrict__ xin,   // [(n+1)*CIN] bf16
              const int*            __restrict__ nbr,   // [n*27] (GATHER only)
              const unsigned short* __restrict__ Wt,    // [KOFF*64*CIN] bf16
              const float*          __restrict__ bias,  // [64]
              int n,
              unsigned short* __restrict__ obf,
              float*          __restrict__ ofl,
              const float*    __restrict__ xres, int col0)
{
    static_assert(KOFF % KC == 0, "KOFF must be a multiple of KC");
    constexpr int NF = CIN / 32;            // WMMA K-steps per kernel offset

    __shared__ __align__(16) unsigned short wlds[KC * 64 * CIN];

    const int tid  = threadIdx.x;
    const int lane = tid & 31;
    const int wave = tid >> 5;
    const int mrow = lane & 15;             // A row / B column within the 16-tile
    const int hi   = lane >> 4;             // half-wave select
    const int base = blockIdx.x * 128 + wave * 16;

    const int r  = base + mrow;                   // point this lane gathers for
    const int rc = (r < n) ? r : (n - 1);         // clamped index for nbr[] read

    const int aoff = hi * 8;    // A: lanes<16 -> K[0..7]&[16..23]; lanes>=16 -> K[8..15]&[24..31]
    const int boff = hi * 8;    // B: lanes<16 -> K[0..7]&[16..23]; lanes>=16 -> K[8..15]&[24..31]

    v8f c0 = {}, c1 = {}, c2 = {}, c3 = {};
    ABFrag a0[NF], a1[NF];

    auto fetchA = [&](int k, ABFrag* dst) {
        int idx;
        if (GATHER) idx = nbr[rc * 27 + k];
        else        idx = rc;
        if (r >= n) idx = n;                      // zero pad row for tail lanes
        const unsigned short* arow = xin + (size_t)idx * CIN;
#pragma unroll
        for (int f = 0; f < NF; ++f) {
            dst[f].u[0] = *(const v4u*)(arow + f * 32 + aoff);
            dst[f].u[1] = *(const v4u*)(arow + f * 32 + 16 + aoff);
        }
    };

    for (int c = 0; c < KOFF / KC; ++c) {
        if (c) __syncthreads();                   // previous chunk fully consumed
        // cooperative copy of this chunk's weights into LDS
        {
            const v4u* src4 = (const v4u*)(Wt + (size_t)c * KC * 64 * CIN);
            v4u*       dst4 = (v4u*)wlds;
            constexpr int NV = KC * 64 * CIN / 8;
#pragma unroll 4
            for (int i = tid; i < NV; i += 256) dst4[i] = src4[i];
        }
        fetchA(c * KC, a0);                       // A loads in flight across barrier
        __syncthreads();

#pragma unroll
        for (int k = 0; k < KC; ++k) {
            ABFrag* cur = (k & 1) ? a1 : a0;
            ABFrag* nxt = (k & 1) ? a0 : a1;
            if (k + 1 < KC) fetchA(c * KC + k + 1, nxt);   // prefetch next offset

            const unsigned short* wk = wlds + (size_t)k * 64 * CIN
                                            + (size_t)mrow * CIN + boff;
#pragma unroll
            for (int f = 0; f < NF; ++f) {
                const unsigned short* wb = wk + f * 32;
                ABFrag b0, b1, b2, b3;
                b0.u[0] = *(const v4u*)(wb);            b0.u[1] = *(const v4u*)(wb + 16);
                b1.u[0] = *(const v4u*)(wb + 16 * CIN); b1.u[1] = *(const v4u*)(wb + 16 * CIN + 16);
                b2.u[0] = *(const v4u*)(wb + 32 * CIN); b2.u[1] = *(const v4u*)(wb + 32 * CIN + 16);
                b3.u[0] = *(const v4u*)(wb + 48 * CIN); b3.u[1] = *(const v4u*)(wb + 48 * CIN + 16);

                c0 = __builtin_amdgcn_wmma_f32_16x16x32_bf16(false, cur[f].bf, false, b0.bf, (short)0, c0, false, false);
                c1 = __builtin_amdgcn_wmma_f32_16x16x32_bf16(false, cur[f].bf, false, b1.bf, (short)0, c1, false, false);
                c2 = __builtin_amdgcn_wmma_f32_16x16x32_bf16(false, cur[f].bf, false, b2.bf, (short)0, c2, false, false);
                c3 = __builtin_amdgcn_wmma_f32_16x16x32_bf16(false, cur[f].bf, false, b3.bf, (short)0, c3, false, false);
            }
        }
    }

    // C/D layout: lane holds column ncol = lane%16; VGPR v is row v + 8*(lane>=16)
    const int ncol = mrow;
    v8f cc[4] = {c0, c1, c2, c3};
#pragma unroll
    for (int t = 0; t < 4; ++t) {
        const float bv = bias[t * 16 + ncol];
#pragma unroll
        for (int v = 0; v < 8; ++v) {
            const int m   = v + 8 * hi;
            const int row = base + m;
            if (row < n) {
                float val = cc[t][v] + bv;
                if (BF16OUT) {
                    val = val > 0.0f ? val : 0.0f;               // ReLU
                    obf[(size_t)row * 64 + t * 16 + ncol] = f2bf(val);
                } else {
                    const size_t o = (size_t)row * 128 + col0 + t * 16 + ncol;
                    ofl[o] = val + xres[o];                       // residual add
                }
            }
        }
    }
}

// ---------- host launch ----------
extern "C" void kernel_launch(void* const* d_in, const int* in_sizes, int n_in,
                              void* d_out, int out_size, void* d_ws, size_t ws_size,
                              hipStream_t stream) {
    const float* x   = (const float*)d_in[0];
    const int*   nbr = (const int*)  d_in[1];
    const float* W00 = (const float*)d_in[2];
    const float* b00 = (const float*)d_in[3];
    const float* W01 = (const float*)d_in[4];
    const float* b01 = (const float*)d_in[5];
    const float* W10 = (const float*)d_in[6];
    const float* b10 = (const float*)d_in[7];
    const float* W11 = (const float*)d_in[8];
    const float* b11 = (const float*)d_in[9];
    const float* W12 = (const float*)d_in[10];
    const float* b12 = (const float*)d_in[11];
    float* out = (float*)d_out;

    const int n = in_sizes[0] / 128;    // N points

    // workspace partition (256B aligned)
    char* ws = (char*)d_ws;
    size_t off = 0;
    auto take = [&](size_t bytes) -> void* {
        off = (off + 255) & ~(size_t)255;
        void* p = ws + off;
        off += bytes;
        return p;
    };
    unsigned short* xpad = (unsigned short*)take((size_t)(n + 1) * 128 * 2);
    unsigned short* h0   = (unsigned short*)take((size_t)(n + 1) * 64 * 2);
    unsigned short* h1   = (unsigned short*)take((size_t)(n + 1) * 64 * 2);
    unsigned short* h1b  = (unsigned short*)take((size_t)(n + 1) * 64 * 2);
    unsigned short* Wt00 = (unsigned short*)take((size_t)27 * 128 * 64 * 2);
    unsigned short* Wt01 = (unsigned short*)take((size_t)27 * 64 * 64 * 2);
    unsigned short* Wt11 = (unsigned short*)take((size_t)27 * 64 * 64 * 2);
    unsigned short* Wt10 = (unsigned short*)take((size_t)128 * 64 * 2);
    unsigned short* Wt12 = (unsigned short*)take((size_t)64 * 64 * 2);
    (void)ws_size; (void)n_in; (void)out_size;

    // ---- prep: convert/transpose + pad rows ----
    const int padtot = (n + 1) * 128;
    k_cvt_x<<<(padtot + 255) / 256, 256, 0, stream>>>(x, xpad, n * 128, padtot);
    k_cvt_w<<<(27 * 128 * 64 + 255) / 256, 256, 0, stream>>>(W00, Wt00, 27, 128);
    k_cvt_w<<<(27 * 64 * 64 + 255) / 256, 256, 0, stream>>>(W01, Wt01, 27, 64);
    k_cvt_w<<<(27 * 64 * 64 + 255) / 256, 256, 0, stream>>>(W11, Wt11, 27, 64);
    k_cvt_w<<<(128 * 64 + 255) / 256, 256, 0, stream>>>(W10, Wt10, 1, 128);
    k_cvt_w<<<(64 * 64 + 255) / 256, 256, 0, stream>>>(W12, Wt12, 1, 64);
    k_zero_pad<<<1, 64, 0, stream>>>(h0 + (size_t)n * 64, h1 + (size_t)n * 64,
                                     h1b + (size_t)n * 64);

    const int blocks = (n + 127) / 128;

    // branch 1 stage 1: h1 = relu(x @ W10 + b10)          (dense, CIN=128)
    k_spconv<128, 1, 1, false, true><<<blocks, 256, 0, stream>>>(
        xpad, nullptr, Wt10, b10, n, h1, nullptr, nullptr, 0);

    // branch 0 stage 1: h0 = relu(sparse_conv(x, W00))     (gather, CIN=128)
    k_spconv<128, 27, 3, true, true><<<blocks, 256, 0, stream>>>(
        xpad, nbr, Wt00, b00, n, h0, nullptr, nullptr, 0);

    // branch 1 stage 2: h1b = relu(sparse_conv(h1, W11))   (gather, CIN=64)
    k_spconv<64, 27, 3, true, true><<<blocks, 256, 0, stream>>>(
        h1, nbr, Wt11, b11, n, h1b, nullptr, nullptr, 0);

    // branch 0 stage 2: out[:, :64] = sparse_conv(h0, W01) + b01 + x[:, :64]
    k_spconv<64, 27, 3, true, false><<<blocks, 256, 0, stream>>>(
        h0, nbr, Wt01, b01, n, nullptr, out, x, 0);

    // branch 1 stage 3: out[:, 64:] = h1b @ W12 + b12 + x[:, 64:]
    k_spconv<64, 1, 1, false, false><<<blocks, 256, 0, stream>>>(
        h1b, nullptr, Wt12, b12, n, nullptr, out, x, 64);
}
